// MultiHeadAttention_3204045603677
// MI455X (gfx1250) — compile-verified, matrix operand order checked by reading, not run on hardware
//
#include <hip/hip_runtime.h>

// ---------------------------------------------------------------------------
// MHA block for MI455X (gfx1250, wave32, WMMA), flash-style softmax.
//   y, attn = LN(Q + (softmax((XQ Wq^T)(XK Wk^T)^T / sqrt(dk)) (XV Wv^T)) Wo^T)
// B=2, S=4096, D=512, H=8, DK=64.
// Data-movement plan: attn (1.07 GB) is written exactly ONCE (final values);
// scores are recomputed in pass B instead of being round-tripped through HBM
// (WMMA recompute ~7us << 3 GB of saved HBM traffic ~130us at 23.3 TB/s).
// ---------------------------------------------------------------------------

#define S_LEN  4096
#define DMODEL 512
#define NH     8
#define DK     64
#define NB     2
#define ROWS   (NB * S_LEN)   // 8192
#define BH     (NB * NH)      // 16

typedef __attribute__((ext_vector_type(16))) __bf16 v16bf;
typedef __attribute__((ext_vector_type(8)))  float  v8f;

__device__ __forceinline__ v8f wmma_bf16(v16bf a, v16bf b, v8f c) {
  // 8 args: (neg_a, A, neg_b, B, c_mod, C, reuse_a, reuse_b)
  return __builtin_amdgcn_wmma_f32_16x16x32_bf16(false, a, false, b, (short)0, c,
                                                 false, false);
}

// Fragment for 16x16x32 bf16 WMMA, per CDNA5 ISA 16-bit A layout:
// lane L holds row (L&15); K elems {kh..kh+7} in elems 0-7 and
// {16+kh..16+kh+7} in elems 8-15, where kh = 8*(L>>4).
// B fragment uses the mirrored mapping (lane holds column), so loading
// row n of W gives column n of B = W^T  (y = x @ W^T pattern).
__device__ __forceinline__ v16bf load_frag_f32(const float* __restrict__ base,
                                               int row, int ld, int kbase, int lane) {
  const float* p = base + (size_t)row * ld + kbase + ((lane >> 4) << 3);
  v16bf f;
#pragma unroll
  for (int j = 0; j < 8; ++j) {
    f[j]     = (__bf16)p[j];
    f[8 + j] = (__bf16)p[16 + j];
  }
  return f;
}

__device__ __forceinline__ v16bf load_frag_bf16(const __bf16* __restrict__ base,
                                                int row, int ld, int kbase, int lane) {
  const __bf16* p = base + (size_t)row * ld + kbase + ((lane >> 4) << 3);
  v16bf f;
#pragma unroll
  for (int j = 0; j < 8; ++j) {
    f[j]     = p[j];
    f[8 + j] = p[16 + j];
  }
  return f;
}

// ---------------------------------------------------------------------------
// K0: projection  out = (X @ W^T + b) * scale,  X:[8192,512] f32, W:[512,512].
// transposed=0 -> out[b,h,s,dk]  [for Q,K];  transposed=1 -> out[b,h,dk,s] [V]
// grid = 2048 blocks * 8 waves = 16384 tiles = (8192/16)*(512/16)
// ---------------------------------------------------------------------------
__global__ void proj_kernel(const float* __restrict__ X, const float* __restrict__ W,
                            const float* __restrict__ bias, __bf16* __restrict__ out,
                            float scale, int transposed) {
  const int lane = threadIdx.x & 31;
  const int wave = threadIdx.x >> 5;
  const int tile = blockIdx.x * 8 + wave;
  const int rbase = (tile >> 5) << 4;
  const int nbase = (tile & 31) << 4;

  v8f acc = {};
#pragma unroll 4
  for (int kc = 0; kc < DMODEL; kc += 32) {
    v16bf a = load_frag_f32(X, rbase + (lane & 15), DMODEL, kc, lane);
    v16bf b = load_frag_f32(W, nbase + (lane & 15), DMODEL, kc, lane);
    acc = wmma_bf16(a, b, acc);
  }
  const int n = nbase + (lane & 15);
  const float bn = bias[n];
  const int h = n >> 6, d = n & 63;
  const int mrow = rbase + ((lane >> 4) << 3);
#pragma unroll
  for (int i = 0; i < 8; ++i) {
    const int r = mrow + i;
    const int bI = r >> 12;
    const int s = r & (S_LEN - 1);
    const float v = (acc[i] + bn) * scale;
    size_t idx;
    if (transposed) idx = ((size_t)(bI * NH + h) * DK + d) * S_LEN + s;
    else            idx = ((size_t)(bI * NH + h) * S_LEN + s) * DK + d;
    out[idx] = (__bf16)v;
  }
}

// ---------------------------------------------------------------------------
// Pass A: softmax stats.  Compute q@k^T tiles (WMMA), masked; per-lane online
// (m, l) over that lane's columns; exact merge: shfl butterfly over the 16
// lanes of each half-wave, then across 8 waves via LDS.  No score is written.
// grid = BH * S/16 = 4096 blocks, 256 threads.
// ---------------------------------------------------------------------------
__global__ void softmax_stats_kernel(const __bf16* __restrict__ qb,
                                     const __bf16* __restrict__ kb,
                                     const int* __restrict__ mask,
                                     float* __restrict__ rowmax,
                                     float* __restrict__ invsum) {
  __shared__ float lmS[8][16];
  __shared__ float llS[8][16];
  const int lane = threadIdx.x & 31, wave = threadIdx.x >> 5;
  const int bh = blockIdx.x >> 8;
  const int qt = blockIdx.x & 255;
  const __bf16* qp = qb + (size_t)bh * S_LEN * DK;
  const __bf16* kp = kb + (size_t)bh * S_LEN * DK;

  const v16bf qa0 = load_frag_bf16(qp, qt * 16 + (lane & 15), DK, 0, lane);
  const v16bf qa1 = load_frag_bf16(qp, qt * 16 + (lane & 15), DK, 32, lane);

  float mi[8], li[8];
#pragma unroll
  for (int i = 0; i < 8; ++i) { mi[i] = -3.0e38f; li[i] = 0.0f; }
  const int mb = (lane >> 4) << 3;

  for (int kt = wave; kt < 256; kt += 8) {
    if (kt + 8 < 256)  // hint the streaming K fragments (global_prefetch_b8)
      __builtin_prefetch(kp + (size_t)(kt + 8) * 16 * DK, 0, 0);
    v8f acc = {};
    v16bf b0 = load_frag_bf16(kp, kt * 16 + (lane & 15), DK, 0, lane);
    v16bf b1 = load_frag_bf16(kp, kt * 16 + (lane & 15), DK, 32, lane);
    acc = wmma_bf16(qa0, b0, acc);
    acc = wmma_bf16(qa1, b1, acc);
    const int kcol = kt * 16 + (lane & 15);
#pragma unroll
    for (int i = 0; i < 8; ++i) {
      const int qrow = qt * 16 + mb + i;
      float sc = acc[i];
      if (mask[(size_t)qrow * S_LEN + kcol] == 0) sc = -1.0e9f;
      const float mn = fmaxf(mi[i], sc);
      li[i] = li[i] * __expf(mi[i] - mn) + __expf(sc - mn);
      mi[i] = mn;
    }
  }
  // merge the 16 lanes of each half-wave (each lane covered a column subset)
#pragma unroll
  for (int msk = 1; msk < 16; msk <<= 1) {
#pragma unroll
    for (int i = 0; i < 8; ++i) {
      const float mo = __shfl_xor(mi[i], msk, 32);
      const float lo = __shfl_xor(li[i], msk, 32);
      const float mn = fmaxf(mi[i], mo);
      li[i] = li[i] * __expf(mi[i] - mn) + lo * __expf(mo - mn);
      mi[i] = mn;
    }
  }
  if ((lane & 15) == 0) {
#pragma unroll
    for (int i = 0; i < 8; ++i) { lmS[wave][mb + i] = mi[i]; llS[wave][mb + i] = li[i]; }
  }
  __syncthreads();
  if (threadIdx.x < 16) {
    float m = lmS[0][threadIdx.x], l = llS[0][threadIdx.x];
#pragma unroll
    for (int w = 1; w < 8; ++w) {
      const float mo = lmS[w][threadIdx.x], lo = llS[w][threadIdx.x];
      const float mn = fmaxf(m, mo);
      l = l * __expf(m - mn) + lo * __expf(mo - mn);
      m = mn;
    }
    const size_t r = (size_t)bh * S_LEN + qt * 16 + threadIdx.x;
    rowmax[r] = m;
    invsum[r] = 1.0f / l;
  }
}

// ---------------------------------------------------------------------------
// Pass B: recompute q@k^T, emit FINAL normalized p = exp(s-m)/l to d_out attn
// (the single HBM pass over the 1 GB attn tensor), stage p as bf16 in LDS to
// re-layout C-fragment (lane=col) -> A-fragment (lane=row), then ctx += p @ v
// with v stored transposed [b,h,dk,s].  ctx comes out fully normalized.
// grid = 4096 blocks, 256 threads; each wave owns 16 K-chunks of 32.
// ---------------------------------------------------------------------------
__global__ void attn_pv_kernel(const __bf16* __restrict__ qb,
                               const __bf16* __restrict__ kb,
                               const __bf16* __restrict__ vtb,
                               const int* __restrict__ mask,
                               const float* __restrict__ rowmax,
                               const float* __restrict__ invsum,
                               float* __restrict__ attn,
                               __bf16* __restrict__ ctxb) {
  __shared__ float  lm[16];
  __shared__ float  li[16];
  __shared__ __bf16 pbuf[8][16][32];       // per-wave p tile, bf16
  __shared__ float  lds_ctx[8][16][64];

  const int lane = threadIdx.x & 31, wave = threadIdx.x >> 5;
  const int bh = blockIdx.x >> 8;
  const int qt = blockIdx.x & 255;
  const int b = bh >> 3, h = bh & 7;

  if (threadIdx.x < 16) {
    const size_t r = (size_t)bh * S_LEN + qt * 16 + threadIdx.x;
    lm[threadIdx.x] = rowmax[r];
    li[threadIdx.x] = invsum[r];
  }
  __syncthreads();

  const __bf16* qp = qb + (size_t)bh * S_LEN * DK;
  const __bf16* kp = kb + (size_t)bh * S_LEN * DK;
  const __bf16* vp = vtb + (size_t)bh * DK * S_LEN;

  const v16bf qa0 = load_frag_bf16(qp, qt * 16 + (lane & 15), DK, 0, lane);
  const v16bf qa1 = load_frag_bf16(qp, qt * 16 + (lane & 15), DK, 32, lane);

  const int mb = (lane >> 4) << 3;
  float mloc[8], nloc[8];
#pragma unroll
  for (int i = 0; i < 8; ++i) { mloc[i] = lm[mb + i]; nloc[i] = li[mb + i]; }

  v8f ctx[4] = {};
  const int r_l = lane & 15;
  const int kh = (lane >> 4) << 3;

  for (int c = wave; c < 128; c += 8) {    // 128 column chunks of 32
    if (c + 8 < 128)
      __builtin_prefetch(kp + (size_t)(c + 8) * 32 * DK, 0, 0);
#pragma unroll
    for (int sub = 0; sub < 2; ++sub) {
      const int kt = c * 2 + sub;
      v8f acc = {};
      v16bf b0 = load_frag_bf16(kp, kt * 16 + r_l, DK, 0, lane);
      v16bf b1 = load_frag_bf16(kp, kt * 16 + r_l, DK, 32, lane);
      acc = wmma_bf16(qa0, b0, acc);
      acc = wmma_bf16(qa1, b1, acc);
      const int kcol = kt * 16 + r_l;
#pragma unroll
      for (int i = 0; i < 8; ++i) {
        const int qrow = qt * 16 + mb + i;
        float sc = acc[i];
        if (mask[(size_t)qrow * S_LEN + kcol] == 0) sc = -1.0e9f;
        const float p = __expf(sc - mloc[i]) * nloc[i];   // final value
        attn[((size_t)bh * S_LEN + qrow) * S_LEN + kcol] = p;
        pbuf[wave][mb + i][sub * 16 + r_l] = (__bf16)p;   // re-layout via LDS
      }
    }
    // A-fragment of p for this 32-wide chunk (lane = row, K-swizzled)
    v16bf pa;
#pragma unroll
    for (int j = 0; j < 8; ++j) {
      pa[j]     = pbuf[wave][r_l][kh + j];
      pa[8 + j] = pbuf[wave][r_l][16 + kh + j];
    }
#pragma unroll
    for (int t = 0; t < 4; ++t) {
      v16bf vb = load_frag_bf16(vp, t * 16 + r_l, S_LEN, c * 32, lane);
      ctx[t] = wmma_bf16(pa, vb, ctx[t]);
    }
  }
  // cross-wave ctx reduction (already normalized)
#pragma unroll
  for (int t = 0; t < 4; ++t)
#pragma unroll
    for (int i = 0; i < 8; ++i) lds_ctx[wave][mb + i][t * 16 + r_l] = ctx[t][i];
  __syncthreads();
  for (int e = threadIdx.x; e < 1024; e += 256) {
    const int r = e >> 6, d = e & 63;
    float s = 0.0f;
#pragma unroll
    for (int w = 0; w < 8; ++w) s += lds_ctx[w][r][d];
    const int srow = qt * 16 + r;
    ctxb[((size_t)(b * S_LEN + srow)) * DMODEL + h * DK + d] = (__bf16)s;
  }
}

// ---------------------------------------------------------------------------
// K4: x = ctx @ Wo^T + bo + residual(Q)   -> xbuf f32 [8192,512]
// ---------------------------------------------------------------------------
__global__ void oproj_kernel(const __bf16* __restrict__ ctxb, const float* __restrict__ Wo,
                             const float* __restrict__ bo, const float* __restrict__ resid,
                             float* __restrict__ xbuf) {
  const int lane = threadIdx.x & 31;
  const int tile = blockIdx.x * 8 + (threadIdx.x >> 5);
  const int rbase = (tile >> 5) << 4;
  const int nbase = (tile & 31) << 4;
  v8f acc = {};
#pragma unroll 4
  for (int kc = 0; kc < DMODEL; kc += 32) {
    v16bf a = load_frag_bf16(ctxb, rbase + (lane & 15), DMODEL, kc, lane);
    v16bf b = load_frag_f32(Wo, nbase + (lane & 15), DMODEL, kc, lane);
    acc = wmma_bf16(a, b, acc);
  }
  const int n = nbase + (lane & 15);
  const float bn = bo[n];
  const int mrow = rbase + ((lane >> 4) << 3);
#pragma unroll
  for (int i = 0; i < 8; ++i) {
    const size_t idx = (size_t)(mrow + i) * DMODEL + n;
    xbuf[idx] = acc[i] + bn + resid[idx];
  }
}

// ---------------------------------------------------------------------------
// K5: LayerNorm over 512, one workgroup (256 threads) per row.
// ---------------------------------------------------------------------------
__global__ void ln_kernel(const float* __restrict__ xbuf, const float* __restrict__ gamma,
                          const float* __restrict__ beta, float* __restrict__ y) {
  __shared__ float red[256];
  const int r = blockIdx.x;
  const float* x = xbuf + (size_t)r * DMODEL;
  const float v0 = x[threadIdx.x], v1 = x[threadIdx.x + 256];
  red[threadIdx.x] = v0 + v1;
  __syncthreads();
  for (int o = 128; o > 0; o >>= 1) {
    if (threadIdx.x < o) red[threadIdx.x] += red[threadIdx.x + o];
    __syncthreads();
  }
  const float mu = red[0] * (1.0f / 512.0f);
  __syncthreads();
  const float d0 = v0 - mu, d1 = v1 - mu;
  red[threadIdx.x] = d0 * d0 + d1 * d1;
  __syncthreads();
  for (int o = 128; o > 0; o >>= 1) {
    if (threadIdx.x < o) red[threadIdx.x] += red[threadIdx.x + o];
    __syncthreads();
  }
  const float rstd = rsqrtf(red[0] * (1.0f / 512.0f) + 1e-6f);
  float* yr = y + (size_t)r * DMODEL;
  yr[threadIdx.x]       = d0 * rstd * gamma[threadIdx.x] + beta[threadIdx.x];
  yr[threadIdx.x + 256] = d1 * rstd * gamma[threadIdx.x + 256] + beta[threadIdx.x + 256];
}

// ---------------------------------------------------------------------------
extern "C" void kernel_launch(void* const* d_in, const int* in_sizes, int n_in,
                              void* d_out, int out_size, void* d_ws, size_t ws_size,
                              hipStream_t stream) {
  const float* Q    = (const float*)d_in[0];
  const float* K    = (const float*)d_in[1];
  const float* V    = (const float*)d_in[2];
  const int*   mask = (const int*)d_in[3];
  const float* w_q  = (const float*)d_in[4];
  const float* b_q  = (const float*)d_in[5];
  const float* w_k  = (const float*)d_in[6];
  const float* b_k  = (const float*)d_in[7];
  const float* w_v  = (const float*)d_in[8];
  const float* b_v  = (const float*)d_in[9];
  const float* w_o  = (const float*)d_in[10];
  const float* b_o  = (const float*)d_in[11];
  const float* g    = (const float*)d_in[12];
  const float* be   = (const float*)d_in[13];

  float* y    = (float*)d_out;                       // [2,4096,512]
  float* attn = y + (size_t)ROWS * DMODEL;           // [2,8,4096,4096]

  char* ws = (char*)d_ws;
  const size_t SZ_QKV = (size_t)BH * S_LEN * DK * 2; // 8 MB each (bf16)
  __bf16* qb     = (__bf16*)(ws);
  __bf16* kb     = (__bf16*)(ws + SZ_QKV);
  __bf16* vtb    = (__bf16*)(ws + 2 * SZ_QKV);       // transposed [b,h,dk,s]
  __bf16* ctxb   = (__bf16*)(ws + 3 * SZ_QKV);       // [B,S,512] bf16
  float*  rowmax = (float*)(ws + 4 * SZ_QKV);        // 256 KB
  float*  invs   = (float*)(ws + 4 * SZ_QKV + (size_t)BH * S_LEN * 4);
  float*  xbuf   = (float*)(ws + 4 * SZ_QKV + (size_t)2 * BH * S_LEN * 4);

  dim3 blk(256);
  proj_kernel<<<2048, blk, 0, stream>>>(Q, w_q, b_q, qb, 0.125f, 0); // fold 1/sqrt(64)
  proj_kernel<<<2048, blk, 0, stream>>>(K, w_k, b_k, kb, 1.0f, 0);
  proj_kernel<<<2048, blk, 0, stream>>>(V, w_v, b_v, vtb, 1.0f, 1);
  softmax_stats_kernel<<<4096, blk, 0, stream>>>(qb, kb, mask, rowmax, invs);
  attn_pv_kernel<<<4096, blk, 0, stream>>>(qb, kb, vtb, mask, rowmax, invs, attn, ctxb);
  oproj_kernel<<<2048, blk, 0, stream>>>(ctxb, w_o, b_o, Q, xbuf);
  ln_kernel<<<8192, blk, 0, stream>>>(xbuf, g, be, y);
}
